// Net_7876970021054
// MI455X (gfx1250) — compile-verified, matrix operand order checked by reading, not run on hardware
//
#include <hip/hip_runtime.h>
#include <hip/hip_bf16.h>

// ---------------------------------------------------------------------------
// 3-layer GCN for MI455X (gfx1250, wave32).
//   h = X @ W          -> bf16 WMMA (v_wmma_f32_16x16x32_bf16), f32 accum
//   g = bias; g[dst] += w_e * h[src]  -> edge-parallel f32 atomics (L2 resident)
//   relu folded into next GEMM's A-load; final log_softmax per node.
// ---------------------------------------------------------------------------

typedef __bf16 bf16_t;
typedef __attribute__((ext_vector_type(16))) __bf16 v16bf;
typedef __attribute__((ext_vector_type(8)))  __bf16 v8bf;
typedef __attribute__((ext_vector_type(8)))  float  v8f;
typedef __attribute__((ext_vector_type(4)))  float  v4f;

__device__ __forceinline__ bf16_t f2bf(float x) { return (bf16_t)x; }

// ---------------------------------------------------------------------------
// Convert weight W[K][Nout] (f32) -> Bt[64][K] (bf16, transposed, zero-padded
// columns Nout..63). Transposed layout makes each WMMA B-fragment load two
// contiguous 16-byte loads per lane.
// ---------------------------------------------------------------------------
__global__ void cvt_weight(const float* __restrict__ W, bf16_t* __restrict__ Bt,
                           int K, int Nout) {
    int i = blockIdx.x * blockDim.x + threadIdx.x;
    if (i >= 64 * K) return;
    int k = i % K;
    int c = i / K;
    float v = (c < Nout) ? W[(size_t)k * Nout + c] : 0.0f;
    Bt[(size_t)c * K + k] = f2bf(v);
}

// ---------------------------------------------------------------------------
// C[M][64] = A[M][K] (f32, optional relu) x Bt^T  (Bt is bf16 [64][K]).
// Block = 128 threads = 4 waves; wave w computes rows [blk*64+w*16, +16) x 64.
// A-fragment layout (16-bit A, 16x32): lanes 0-15 hold K = kb+{0..7,16..23},
// lanes 16-31 hold K = kb+{8..15,24..31}; B symmetric per column.
// ---------------------------------------------------------------------------
template <int K, bool RELU>
__global__ __launch_bounds__(128) void gemm_wmma_bf16(
    const float* __restrict__ A, const bf16_t* __restrict__ Bt,
    float* __restrict__ C, int M) {
    const int wave   = threadIdx.x >> 5;
    const int lane   = threadIdx.x & 31;
    const int laneLo = lane & 15;
    const int laneHi = lane >> 4;
    const int rowBase = blockIdx.x * 64 + wave * 16;

    int arow = rowBase + laneLo;
    if (arow >= M) arow = M - 1;          // clamp: keeps EXEC all-ones for WMMA
    const float* __restrict__ ap = A + (size_t)arow * K;

    v8f acc[4];
#pragma unroll
    for (int t = 0; t < 4; ++t) acc[t] = (v8f){0.f,0.f,0.f,0.f,0.f,0.f,0.f,0.f};

#pragma unroll 4
    for (int kb = 0; kb < K; kb += 32) {
        // ---- A fragment: 4x contiguous float4 loads, convert to bf16 ----
        const float* a0 = ap + kb + laneHi * 8;
        v4f f0 = *(const v4f*)(a0);
        v4f f1 = *(const v4f*)(a0 + 4);
        v4f f2 = *(const v4f*)(a0 + 16);
        v4f f3 = *(const v4f*)(a0 + 20);
        if (RELU) {
#pragma unroll
            for (int j = 0; j < 4; ++j) {
                f0[j] = fmaxf(f0[j], 0.f);
                f1[j] = fmaxf(f1[j], 0.f);
                f2[j] = fmaxf(f2[j], 0.f);
                f3[j] = fmaxf(f3[j], 0.f);
            }
        }
        v16bf a;
#pragma unroll
        for (int j = 0; j < 4; ++j) {
            a[j]      = f2bf(f0[j]);
            a[4 + j]  = f2bf(f1[j]);
            a[8 + j]  = f2bf(f2[j]);
            a[12 + j] = f2bf(f3[j]);
        }

        // ---- 4 B fragments (16 columns each) + WMMA ----
#pragma unroll
        for (int t = 0; t < 4; ++t) {
            const bf16_t* bp = Bt + (size_t)(t * 16 + laneLo) * K + kb + laneHi * 8;
            v8bf b0 = *(const v8bf*)(bp);
            v8bf b1 = *(const v8bf*)(bp + 16);
            v16bf b = __builtin_shufflevector(b0, b1,
                0, 1, 2, 3, 4, 5, 6, 7, 8, 9, 10, 11, 12, 13, 14, 15);
            acc[t] = __builtin_amdgcn_wmma_f32_16x16x32_bf16(
                false, a, false, b, (short)0, acc[t], false, false);
        }
    }

    // ---- store D: VGPR v -> row rowBase + v + laneHi*8, col t*16 + laneLo ----
#pragma unroll
    for (int t = 0; t < 4; ++t) {
#pragma unroll
        for (int v = 0; v < 8; ++v) {
            int r = rowBase + v + laneHi * 8;
            if (r < M) C[(size_t)r * 64 + t * 16 + laneLo] = acc[t][v];
        }
    }
}

// out[n][d] = bias[d]   (seed for scatter-add)
__global__ void init_bias(float* __restrict__ out, const float* __restrict__ b,
                          int total, int D) {
    int i = blockIdx.x * blockDim.x + threadIdx.x;
    if (i >= total) return;
    out[i] = b[i % D];
}

// g[dst[e]][q*4..+3] += w[e] * h[src[e]][q*4..+3]
// 16 (or 10) lanes per edge -> coalesced 16B gathers; atomics land in L2
// (h is 25.6 MB, fits the 192 MB L2).
__global__ void scatter_edges(const int* __restrict__ src,
                              const int* __restrict__ dst,
                              const float* __restrict__ ew,
                              const float* __restrict__ h, int ldH,
                              float* __restrict__ out, int D, int total) {
    int i = blockIdx.x * blockDim.x + threadIdx.x;
    if (i >= total) return;
    int q4 = D >> 2;
    int e = i / q4;
    int q = i - e * q4;
    int s = src[e];
    int d = dst[e];
    float w = ew[e];
    v4f hv = *(const v4f*)(h + (size_t)s * ldH + q * 4);
    float* ob = out + (size_t)d * D + q * 4;
    atomicAdd(ob + 0, w * hv[0]);
    atomicAdd(ob + 1, w * hv[1]);
    atomicAdd(ob + 2, w * hv[2]);
    atomicAdd(ob + 3, w * hv[3]);
}

// row-wise log_softmax over 40 classes
__global__ void log_softmax40(const float* __restrict__ x,
                              float* __restrict__ out, int N) {
    int n = blockIdx.x * blockDim.x + threadIdx.x;
    if (n >= N) return;
    const float* xr = x + (size_t)n * 40;
    float m = -3.4028235e38f;
#pragma unroll
    for (int j = 0; j < 40; ++j) m = fmaxf(m, xr[j]);
    float s = 0.f;
#pragma unroll
    for (int j = 0; j < 40; ++j) s += __expf(xr[j] - m);
    float ls = __logf(s);
#pragma unroll
    for (int j = 0; j < 40; ++j) out[(size_t)n * 40 + j] = xr[j] - m - ls;
}

static inline size_t align_up(size_t x) { return (x + 255) & ~(size_t)255; }

extern "C" void kernel_launch(void* const* d_in, const int* in_sizes, int n_in,
                              void* d_out, int out_size, void* d_ws, size_t ws_size,
                              hipStream_t stream) {
    const int*   edge_index = (const int*)d_in[0];     // [2][E]
    const float* features   = (const float*)d_in[1];   // [N][512]
    const float* ew         = (const float*)d_in[2];   // [E]
    const float* W1 = (const float*)d_in[3];           // [512][64]
    const float* b1 = (const float*)d_in[4];
    const float* W2 = (const float*)d_in[5];           // [64][64]
    const float* b2 = (const float*)d_in[6];
    const float* W3 = (const float*)d_in[7];           // [64][40]
    const float* b3 = (const float*)d_in[8];
    float* out = (float*)d_out;

    const int E = in_sizes[0] / 2;
    const int N = in_sizes[1] / 512;
    const int* src = edge_index;
    const int* dst = edge_index + E;

    // ---- workspace carve-up ----
    char* p = (char*)d_ws;
    size_t off = 0;
    bf16_t* Wb1 = (bf16_t*)(p + off); off = align_up(off + (size_t)64 * 512 * sizeof(bf16_t));
    bf16_t* Wb2 = (bf16_t*)(p + off); off = align_up(off + (size_t)64 * 64  * sizeof(bf16_t));
    bf16_t* Wb3 = (bf16_t*)(p + off); off = align_up(off + (size_t)64 * 64  * sizeof(bf16_t));
    float* h1 = (float*)(p + off); off = align_up(off + (size_t)N * 64 * sizeof(float));
    float* g1 = (float*)(p + off); off = align_up(off + (size_t)N * 64 * sizeof(float));
    float* h2 = (float*)(p + off); off = align_up(off + (size_t)N * 64 * sizeof(float));
    float* g2 = (float*)(p + off); off = align_up(off + (size_t)N * 64 * sizeof(float));
    float* h3 = (float*)(p + off); off = align_up(off + (size_t)N * 64 * sizeof(float));
    float* g3 = (float*)(p + off); off = align_up(off + (size_t)N * 40 * sizeof(float));
    (void)ws_size; (void)n_in; (void)out_size;

    const int TB = 256;
    const int gemmBlocks = (N + 63) / 64;

    // weight conversions (tiny)
    cvt_weight<<<(64 * 512 + TB - 1) / TB, TB, 0, stream>>>(W1, Wb1, 512, 64);
    cvt_weight<<<(64 * 64  + TB - 1) / TB, TB, 0, stream>>>(W2, Wb2, 64, 64);
    cvt_weight<<<(64 * 64  + TB - 1) / TB, TB, 0, stream>>>(W3, Wb3, 64, 40);

    // ---- layer 1: h1 = X @ W1 ; g1 = b1 + scatter(w * h1[src]) ----
    gemm_wmma_bf16<512, false><<<gemmBlocks, 128, 0, stream>>>(features, Wb1, h1, N);
    {
        int tot = N * 64;
        init_bias<<<(tot + TB - 1) / TB, TB, 0, stream>>>(g1, b1, tot, 64);
        int etot = E * 16;  // D=64 -> 16 quads per edge
        scatter_edges<<<(etot + TB - 1) / TB, TB, 0, stream>>>(src, dst, ew, h1, 64, g1, 64, etot);
    }

    // ---- layer 2: h2 = relu(g1) @ W2 ; g2 = b2 + scatter ----
    gemm_wmma_bf16<64, true><<<gemmBlocks, 128, 0, stream>>>(g1, Wb2, h2, N);
    {
        int tot = N * 64;
        init_bias<<<(tot + TB - 1) / TB, TB, 0, stream>>>(g2, b2, tot, 64);
        int etot = E * 16;
        scatter_edges<<<(etot + TB - 1) / TB, TB, 0, stream>>>(src, dst, ew, h2, 64, g2, 64, etot);
    }

    // ---- layer 3: h3 = relu(g2) @ W3 (cols 40..63 are zero) ; g3 = b3 + scatter ----
    gemm_wmma_bf16<64, true><<<gemmBlocks, 128, 0, stream>>>(g2, Wb3, h3, N);
    {
        int tot = N * 40;
        init_bias<<<(tot + TB - 1) / TB, TB, 0, stream>>>(g3, b3, tot, 40);
        int etot = E * 10;  // D=40 -> 10 quads per edge
        scatter_edges<<<(etot + TB - 1) / TB, TB, 0, stream>>>(src, dst, ew, h3, 64, g3, 40, etot);
    }

    // ---- log_softmax ----
    log_softmax40<<<(N + TB - 1) / TB, TB, 0, stream>>>(g3, out, N);
}
